// SparseBEVSelfAttention_12446815224515
// MI455X (gfx1250) — matrix operand order repaired and checked by reading, not compiled, not run on hardware
//
#include <hip/hip_runtime.h>
#include <hip/hip_bf16.h>
#include <math.h>

// SparseBEV self-attention, MI455X (gfx1250, wave32, WMMA 16x16).
// - Flash-fused distance-biased attention (never materialize [H,Q,K])
// - bf16 WMMA for P·V and for the proj/MLP GEMM chain, f32 LN/softmax math
// - TDM (tensor_load_to_lds) double-buffers the mask stream (the only real
//   HBM traffic, 67MB) into LDS, overlapped with compute via TENSORcnt.
// - exp2-domain online softmax (raw v_exp_f32) with per-tile row max/min
//   precomputed during tile prep -> one rescale + one reduce per K-tile.

#define QN 4096
#define CN 256
#define HN 8
#define DN 32

typedef __attribute__((ext_vector_type(16))) __bf16 v16bf;
typedef __attribute__((ext_vector_type(8)))  float  v8f;
typedef __attribute__((ext_vector_type(4)))  unsigned int v4u;
typedef __attribute__((ext_vector_type(8)))  int  v8i_t;
typedef __attribute__((ext_vector_type(4)))  int  v4i_t;

union BF16x16 { v16bf v; uint4 q[2]; unsigned short u[16]; };

#ifndef __has_builtin
#define __has_builtin(x) 0
#endif

#if __has_builtin(__builtin_amdgcn_tensor_load_to_lds) && __has_builtin(__builtin_amdgcn_s_wait_tensorcnt)
#define USE_TDM 1
#warning "CDNA5 path report: TDM tensor_load_to_lds ENABLED"
#else
#define USE_TDM 0
#warning "CDNA5 path report: TDM builtin NOT available - plain VMEM mask loads"
#endif

#if __has_builtin(__builtin_amdgcn_exp2f)
#define EXP2(x) __builtin_amdgcn_exp2f(x)
#warning "CDNA5 path report: raw v_exp_f32 via __builtin_amdgcn_exp2f"
#else
#define EXP2(x) exp2f(x)
#warning "CDNA5 path report: libm exp2f fallback"
#endif

#if __has_builtin(__builtin_amdgcn_cvt_pk_bf16_f32)
#define HAVE_PK_BF16 1
#warning "CDNA5 path report: v_cvt_pk_bf16_f32 ENABLED"
#else
#define HAVE_PK_BF16 0
#warning "CDNA5 path report: scalar bf16 round/pack fallback"
#endif

static __device__ __forceinline__ unsigned short f2bf(float f) {
  unsigned int u = __float_as_uint(f);
  u += 0x7FFFu + ((u >> 16) & 1u);       // round-to-nearest-even
  return (unsigned short)(u >> 16);
}

// Pack two f32 -> two bf16 (RNE) into 4 bytes at dst.
static __device__ __forceinline__ void pack2bf(unsigned short* dst, float a, float b) {
#if HAVE_PK_BF16
  auto pk = __builtin_amdgcn_cvt_pk_bf16_f32(a, b);
  __builtin_memcpy(dst, &pk, 4);
#else
  dst[0] = f2bf(a);
  dst[1] = f2bf(b);
#endif
}

// B-matrix 32x16 bf16 fragment: lanes 0-15 hold K=0..7 & 16..23, lanes 16-31
// hold K=8..15 & 24..31 (same K order as the A fragment).
static __device__ __forceinline__ v16bf load_b_frag(const unsigned short* p, int hi) {
  BF16x16 t;
  t.q[0] = *(const uint4*)(p + hi * 8);
  t.q[1] = *(const uint4*)(p + hi * 8 + 16);
  return t.v;
}
// A-matrix 16x32 bf16 fragment: lanes 0-15 hold K=0..7 & 16..23,
// lanes 16-31 hold K=8..15 & 24..31 (ISA 7.12.2 table).
static __device__ __forceinline__ v16bf load_a_frag(const unsigned short* p, int hi) {
  BF16x16 t;
  t.q[0] = *(const uint4*)(p + hi * 8);
  t.q[1] = *(const uint4*)(p + hi * 8 + 16);
  return t.v;
}

#if USE_TDM
// Issue a TDM load of a 16x128 i32 tile of the [4096x4096] mask tensor into
// LDS at lds_byte_addr.  D# layout per CDNA5 ISA ch.8 (2D tensor, groups 2/3
// zero).  One wave issues it; EXEC is ignored by tensor ops.
static __device__ __forceinline__ void tdm_load_mask_tile(const int* gsrc, unsigned lds_byte_addr) {
  unsigned long long ga = (unsigned long long)(const void*)gsrc;
  v4u g0;
  g0[0] = 1u;                                            // count=1 (valid user D#)
  g0[1] = lds_byte_addr;                                 // lds_addr [63:32]
  g0[2] = (unsigned)(ga & 0xFFFFFFFFull);                // global_addr [95:64]
  g0[3] = (unsigned)((ga >> 32) & 0x01FFFFFFull)         // global_addr [120:96]
        | (2u << 30);                                    // type=2 ("image") [127:126]
  v8i_t g1;
  g1[0] = 0x20000;                                       // data_size=2 (4 bytes)
  g1[1] = (int)(4096u << 16);                            // tensor_dim0=4096 (lo16 @ [63:48])
  g1[2] = (int)(4096u << 16);                            // dim0 hi16=0 | tensor_dim1=4096 lo16
  g1[3] = (int)(128u << 16);                             // dim1 hi16=0 | tile_dim0=128
  g1[4] = 16;                                            // tile_dim1=16, tile_dim2=0
  g1[5] = 4096;                                          // tensor_dim0_stride lo32
  g1[6] = 0;                                             // stride hi16 | dim1_stride lo16
  g1[7] = 0;
  v4i_t g2 = {0, 0, 0, 0};
  v4i_t g3 = {0, 0, 0, 0};
#if defined(__clang_major__) && __clang_major__ >= 23
  v8i_t g4 = {0, 0, 0, 0, 0, 0, 0, 0};
  __builtin_amdgcn_tensor_load_to_lds(g0, g1, g2, g3, g4, 0);
#else
  __builtin_amdgcn_tensor_load_to_lds(g0, g1, g2, g3, 0);
#endif
}
#endif

// ---------------------------------------------------------------------------
// Kernel 1: per-row prep.  vp = LN(v)*Wv^T + bv (f32), tau = q*Wtau^T + btau.
// ---------------------------------------------------------------------------
__global__ __launch_bounds__(256) void prep_kernel(
    const float* __restrict__ vIn, const float* __restrict__ qIn,
    const float* __restrict__ g, const float* __restrict__ b,
    const float* __restrict__ Wv, const float* __restrict__ bv,
    const float* __restrict__ Wtau, const float* __restrict__ btau,
    float* __restrict__ vp, float* __restrict__ tau)
{
  const int q = blockIdx.x, t = threadIdx.x;
  const int lane = t & 31, wave = t >> 5;
  __shared__ __align__(16) float sv[CN];
  __shared__ __align__(16) float sq[CN];
  __shared__ float red[16];

  float x = vIn[(size_t)q * CN + t];
  sq[t] = qIn[(size_t)q * CN + t];
  float xs = x, xq = x * x;
  #pragma unroll
  for (int off = 16; off; off >>= 1) {
    xs += __shfl_xor(xs, off, 32);
    xq += __shfl_xor(xq, off, 32);
  }
  if (lane == 0) { red[wave] = xs; red[8 + wave] = xq; }
  __syncthreads();
  float sm = 0.f, ss = 0.f;
  #pragma unroll
  for (int i = 0; i < 8; ++i) { sm += red[i]; ss += red[8 + i]; }
  const float mu  = sm * (1.f / CN);
  const float rsd = rsqrtf(ss * (1.f / CN) - mu * mu + 1e-5f);
  sv[t] = (x - mu) * rsd * g[t] + b[t];
  __syncthreads();

  float acc = bv[t];
  const float4* w4 = (const float4*)(Wv + (size_t)t * CN);
  const float4* v4 = (const float4*)sv;
  #pragma unroll 8
  for (int k = 0; k < CN / 4; ++k) {
    float4 wv = w4[k], vv = v4[k];
    acc += wv.x * vv.x + wv.y * vv.y + wv.z * vv.z + wv.w * vv.w;
  }
  vp[(size_t)q * CN + t] = acc;

  if (t < HN) {
    float a2 = btau[t];
    const float4* wt = (const float4*)(Wtau + t * CN);
    const float4* q4 = (const float4*)sq;
    for (int k = 0; k < CN / 4; ++k) {
      float4 wv = wt[k], qv = q4[k];
      a2 += wv.x * qv.x + wv.y * qv.y + wv.z * qv.z + wv.w * qv.w;
    }
    tau[q * HN + t] = a2;
  }
}

// ---------------------------------------------------------------------------
// Kernel 2: convert GEMM weights to bf16 ([out][in] layout: B-fragment lane
// (N = out col) reads a contiguous K run).
// ---------------------------------------------------------------------------
__global__ __launch_bounds__(256) void cvtw_kernel(
    const float* __restrict__ Wp, const float* __restrict__ Wm1,
    const float* __restrict__ Wm2, unsigned short* __restrict__ WpB,
    unsigned short* __restrict__ Wm1B, unsigned short* __restrict__ Wm2B)
{
  int idx = blockIdx.x * 256 + threadIdx.x;
  if (idx < 65536)       WpB[idx]           = f2bf(Wp[idx]);
  else if (idx < 196608) Wm1B[idx - 65536]  = f2bf(Wm1[idx - 65536]);
  else if (idx < 327680) Wm2B[idx - 196608] = f2bf(Wm2[idx - 196608]);
}

// ---------------------------------------------------------------------------
// Kernel 3: transpose vp[q][c] f32 -> vpT[c][q] bf16 (B-side of P·V).
// ---------------------------------------------------------------------------
__global__ __launch_bounds__(256) void transpose_kernel(
    const float* __restrict__ vp, unsigned short* __restrict__ vpT)
{
  __shared__ float tile[32][33];
  const int t = threadIdx.x, tx = t & 31, ty = t >> 5;
  const int q0 = blockIdx.x * 32, c0 = blockIdx.y * 32;
  #pragma unroll
  for (int i = 0; i < 4; ++i) {
    int qq = ty + 8 * i;
    tile[qq][tx] = vp[(size_t)(q0 + qq) * CN + c0 + tx];
  }
  __syncthreads();
  #pragma unroll
  for (int i = 0; i < 4; ++i) {
    int cc = ty + 8 * i;
    vpT[(size_t)(c0 + cc) * QN + q0 + tx] = f2bf(tile[tx][cc]);
  }
}

// ---------------------------------------------------------------------------
// Kernel 4: flash attention.  Block = 16 queries x 8 heads (wave == head).
// Per 128-K tile: TDM-staged mask + cooperative dist tile in LDS, per-row
// tile max/min, then 4 WMMA K-steps of exp2-domain online softmax + P·V.
// ---------------------------------------------------------------------------
__global__ __launch_bounds__(256) void flash_kernel(
    const float* __restrict__ qpos, const int* __restrict__ mask,
    const float* __restrict__ tau, const unsigned short* __restrict__ vpT,
    unsigned short* __restrict__ aOut)
{
  const int tid = threadIdx.x;
  const int lane = tid & 31, wave = tid >> 5;   // wave == head h
  const int laneM = lane & 15, hi = lane >> 4;
  const int q0 = blockIdx.x * 16;

  __shared__ __align__(16) float sdist[16][132];
  __shared__ __align__(16) float sadd[16][132];
  __shared__ __align__(16) int   smaskraw[2][16][128];   // TDM double buffer
  __shared__ float spmax[16][16];
  __shared__ float spmin[16][16];
  __shared__ float srmax[16];
  __shared__ float srmin[16];
  __shared__ float stau[16][8];
  __shared__ float sqc[16][2];

  if (tid < 128) stau[tid >> 3][tid & 7] = tau[(q0 + (tid >> 3)) * HN + (tid & 7)];
  if (tid < 16) {
    float ux = qpos[(q0 + tid) * 2 + 0], uy = qpos[(q0 + tid) * 2 + 1];
    sqc[tid][0] = ux * 102.4f - 51.2f;
    sqc[tid][1] = uy * 102.4f - 51.2f;
  }

#if USE_TDM
  if (wave == 0) {   // stage tile 0
    tdm_load_mask_tile(mask + (size_t)q0 * QN,
                       (unsigned)(size_t)(void*)&smaskraw[0][0][0]);
  }
#endif
  __syncthreads();

  // exp2-domain: fold log2(e) into tau once.
  const float tauL2 = stau[laneM][wave] * 1.44269504089f;
  const int pr = tid >> 4;            // prep: row 0..15
  const int pcw = tid & 15;           // prep: partial-reduction slot
  const int pc = pcw * 8;             // prep: 8 contiguous cols
  const float prx = sqc[pr][0], pry = sqc[pr][1];

  float m = -1e30f, l = 0.f;
  v8f acc0 = {}; v8f acc1 = {};
  const unsigned short* brow0 = vpT + (size_t)(wave * DN + laneM) * QN;
  const unsigned short* brow1 = brow0 + (size_t)16 * QN;

  for (int t = 0; t < QN / 128; ++t) {
    const int k0 = t * 128;
    const int buf = t & 1;
    __syncthreads();                  // (A) previous consumers done
#if USE_TDM
    if (wave == 0) {
      if (t + 1 < QN / 128) {
        tdm_load_mask_tile(mask + (size_t)q0 * QN + (k0 + 128),
                           (unsigned)(size_t)(void*)&smaskraw[buf ^ 1][0][0]);
        __builtin_amdgcn_s_wait_tensorcnt(1);   // tile t resident
      } else {
        __builtin_amdgcn_s_wait_tensorcnt(0);
      }
    }
    __syncthreads();                  // (B) mask tile t visible to all waves
#endif
    // cooperative dist/mask tile + per-row unmasked max/min partials
    {
      float pmx = -1e30f, pmn = 1e30f;
      const float2* kp = (const float2*)qpos + (k0 + pc);
#if !USE_TDM
      const int* mrow = mask + (size_t)(q0 + pr) * QN + (k0 + pc);
#endif
      #pragma unroll
      for (int i = 0; i < 8; ++i) {
#if USE_TDM
        int mv = smaskraw[buf][pr][pc + i];
#else
        int mv = mrow[i];
#endif
        float2 ku = kp[i];
        float kx = ku.x * 102.4f - 51.2f;
        float ky = ku.y * 102.4f - 51.2f;
        float dx = prx - kx, dy = pry - ky;
        float d = -sqrtf(dx * dx + dy * dy);
        sdist[pr][pc + i] = d;
        sadd[pr][pc + i]  = mv ? -1e30f : 0.f;
        if (!mv) { pmx = fmaxf(pmx, d); pmn = fminf(pmn, d); }
      }
      spmax[pr][pcw] = pmx;
      spmin[pr][pcw] = pmn;
    }
    __syncthreads();                  // (C) partials ready
    if (tid < 16) {
      float mx = spmax[tid][0], mn = spmin[tid][0];
      #pragma unroll
      for (int j = 1; j < 16; ++j) {
        mx = fmaxf(mx, spmax[tid][j]);
        mn = fminf(mn, spmin[tid][j]);
      }
      srmax[tid] = mx; srmin[tid] = mn;
    }
    __syncthreads();                  // (D) row stats ready

    // per-row exact tile max of s = tauL2*d (+0 for unmasked)
    const float rmx = srmax[laneM], rmn = srmin[laneM];
    const float mt = (rmx == -1e30f) ? -1e30f
                   : ((tauL2 >= 0.f) ? tauL2 * rmx : tauL2 * rmn);
    const float mnew = fmaxf(m, mt);
    const float rsc = EXP2(m - mnew);
    m = mnew;
    l *= rsc;
    // rescale accumulator rows once per tile (C layout: VGPR v -> rows v/v+8)
    #pragma unroll
    for (int vv = 0; vv < 8; ++vv) {
      float rl = __shfl(rsc, vv, 32);
      float rh = __shfl(rsc, vv + 8, 32);
      float sc = hi ? rh : rl;
      acc0[vv] *= sc; acc1[vv] *= sc;
    }

    float ps = 0.f;
    #pragma unroll
    for (int j = 0; j < 4; ++j) {
      const int kb = 32 * j + hi * 8;
      const float* dr = &sdist[laneM][kb];
      const float* ar = &sadd[laneM][kb];
      BF16x16 af;
      #pragma unroll
      for (int i = 0; i < 4; ++i) {
        float p0 = EXP2(fmaf(dr[2*i],        tauL2, ar[2*i])        - m);
        float p1 = EXP2(fmaf(dr[2*i + 1],    tauL2, ar[2*i + 1])    - m);
        float p2 = EXP2(fmaf(dr[2*i + 16],   tauL2, ar[2*i + 16])   - m);
        float p3 = EXP2(fmaf(dr[2*i + 17],   tauL2, ar[2*i + 17])   - m);
        ps += (p0 + p1) + (p2 + p3);
        pack2bf(&af.u[2*i],     p0, p1);
        pack2bf(&af.u[2*i + 8], p2, p3);
      }
      const int kk = k0 + 32 * j;
      v16bf b0 = load_b_frag(brow0 + kk, hi);
      v16bf b1 = load_b_frag(brow1 + kk, hi);
      acc0 = __builtin_amdgcn_wmma_f32_16x16x32_bf16(false, af.v, false, b0, (short)0, acc0, false, false);
      acc1 = __builtin_amdgcn_wmma_f32_16x16x32_bf16(false, af.v, false, b1, (short)0, acc1, false, false);
    }
    ps += __shfl_xor(ps, 16, 32);     // one cross-lane reduce per tile
    l += ps;
  }

  #pragma unroll
  for (int vv = 0; vv < 8; ++vv) {
    float ll = __shfl(l, vv, 32);
    float lh = __shfl(l, vv + 8, 32);
    float inv = 1.0f / (hi ? lh : ll);
    int row = vv + hi * 8;
    size_t base = (size_t)(q0 + row) * CN + wave * DN + laneM;
    aOut[base]      = f2bf(acc0[vv] * inv);
    aOut[base + 16] = f2bf(acc1[vv] * inv);
  }
}

// ---------------------------------------------------------------------------
// Kernel 5: fused epilogue per 16-row tile:
//   z = a*Wp^T + bp -> LN -> h = GELU(z*Wm1^T+bm1)*Wm2^T+bm2 -> LN(z+h) -> out
// ---------------------------------------------------------------------------
__global__ __launch_bounds__(256) void epilogue_kernel(
    const unsigned short* __restrict__ aB,
    const unsigned short* __restrict__ WpB, const float* __restrict__ bproj,
    const float* __restrict__ gpre, const float* __restrict__ bpre,
    const unsigned short* __restrict__ Wm1B, const float* __restrict__ bm1,
    const unsigned short* __restrict__ Wm2B, const float* __restrict__ bm2,
    const float* __restrict__ gpost, const float* __restrict__ bpost,
    float* __restrict__ out)
{
  const int tid = threadIdx.x;
  const int lane = tid & 31, wave = tid >> 5;
  const int laneM = lane & 15, hi = lane >> 4;
  const int q0 = blockIdx.x * 16;

  __shared__ __align__(16) unsigned short ab[16][264];
  __shared__ __align__(16) unsigned short zb[16][264];
  __shared__ __align__(16) unsigned short hb[16][520];
  __shared__ __align__(16) float zf[16][260];

  {
    int row = tid >> 4, c16 = (tid & 15) * 16;
    const unsigned short* src = aB + (size_t)(q0 + row) * CN + c16;
    *(uint4*)&ab[row][c16]     = *(const uint4*)(src);
    *(uint4*)&ab[row][c16 + 8] = *(const uint4*)(src + 8);
  }
  __syncthreads();

  // proj: 16 N-tiles over 8 waves
  #pragma unroll
  for (int tt = 0; tt < 2; ++tt) {
    const int col = (wave * 2 + tt) * 16 + laneM;
    const unsigned short* brow = WpB + (size_t)col * CN;
    const unsigned short* arow = &ab[laneM][0];
    v8f c = {};
    #pragma unroll
    for (int jk = 0; jk < 8; ++jk)
      c = __builtin_amdgcn_wmma_f32_16x16x32_bf16(false, load_a_frag(arow + 32 * jk, hi),
                                                  false, load_b_frag(brow + 32 * jk, hi),
                                                  (short)0, c, false, false);
    const float bias = bproj[col];
    #pragma unroll
    for (int vv = 0; vv < 8; ++vv) zf[vv + hi * 8][col] = c[vv] + bias;
  }
  __syncthreads();

  // LN pre (2 rows per wave)
  #pragma unroll
  for (int rr = 0; rr < 2; ++rr) {
    const int row = wave * 2 + rr;
    float vals[8]; float sm = 0.f, sq = 0.f;
    #pragma unroll
    for (int i = 0; i < 8; ++i) { float x = zf[row][lane + 32 * i]; vals[i] = x; sm += x; sq += x * x; }
    #pragma unroll
    for (int off = 16; off; off >>= 1) { sm += __shfl_xor(sm, off, 32); sq += __shfl_xor(sq, off, 32); }
    float mu = sm * (1.f / CN);
    float rstd = rsqrtf(sq * (1.f / CN) - mu * mu + 1e-5f);
    #pragma unroll
    for (int i = 0; i < 8; ++i) {
      int cc = lane + 32 * i;
      float y = (vals[i] - mu) * rstd * gpre[cc] + bpre[cc];
      zf[row][cc] = y; zb[row][cc] = f2bf(y);
    }
  }
  __syncthreads();

  // m1 + exact GELU: 32 N-tiles over 8 waves
  #pragma unroll
  for (int tt = 0; tt < 4; ++tt) {
    const int col = (wave * 4 + tt) * 16 + laneM;
    const unsigned short* brow = Wm1B + (size_t)col * CN;
    const unsigned short* arow = &zb[laneM][0];
    v8f c = {};
    #pragma unroll
    for (int jk = 0; jk < 8; ++jk)
      c = __builtin_amdgcn_wmma_f32_16x16x32_bf16(false, load_a_frag(arow + 32 * jk, hi),
                                                  false, load_b_frag(brow + 32 * jk, hi),
                                                  (short)0, c, false, false);
    const float bias = bm1[col];
    #pragma unroll
    for (int vv = 0; vv < 8; ++vv) {
      float x = c[vv] + bias;
      hb[vv + hi * 8][col] = f2bf(0.5f * x * (1.0f + erff(x * 0.70710678f)));
    }
  }
  __syncthreads();

  // m2 + residual
  #pragma unroll
  for (int tt = 0; tt < 2; ++tt) {
    const int col = (wave * 2 + tt) * 16 + laneM;
    const unsigned short* brow = Wm2B + (size_t)col * (2 * CN);
    const unsigned short* arow = &hb[laneM][0];
    v8f c = {};
    #pragma unroll
    for (int jk = 0; jk < 16; ++jk)
      c = __builtin_amdgcn_wmma_f32_16x16x32_bf16(false, load_a_frag(arow + 32 * jk, hi),
                                                  false, load_b_frag(brow + 32 * jk, hi),
                                                  (short)0, c, false, false);
    const float bias = bm2[col];
    #pragma unroll
    for (int vv = 0; vv < 8; ++vv) {
      int row = vv + hi * 8;
      zf[row][col] = c[vv] + bias + zf[row][col];
    }
  }
  __syncthreads();

  // LN post -> out
  #pragma unroll
  for (int rr = 0; rr < 2; ++rr) {
    const int row = wave * 2 + rr;
    float vals[8]; float sm = 0.f, sq = 0.f;
    #pragma unroll
    for (int i = 0; i < 8; ++i) { float x = zf[row][lane + 32 * i]; vals[i] = x; sm += x; sq += x * x; }
    #pragma unroll
    for (int off = 16; off; off >>= 1) { sm += __shfl_xor(sm, off, 32); sq += __shfl_xor(sq, off, 32); }
    float mu = sm * (1.f / CN);
    float rstd = rsqrtf(sq * (1.f / CN) - mu * mu + 1e-5f);
    #pragma unroll
    for (int i = 0; i < 8; ++i) {
      int cc = lane + 32 * i;
      out[(size_t)(q0 + row) * CN + cc] = (vals[i] - mu) * rstd * gpost[cc] + bpost[cc];
    }
  }
}

// ---------------------------------------------------------------------------
extern "C" void kernel_launch(void* const* d_in, const int* in_sizes, int n_in,
                              void* d_out, int out_size, void* d_ws, size_t ws_size,
                              hipStream_t stream) {
  (void)in_sizes; (void)n_in; (void)out_size; (void)ws_size;
  const float* qpos = (const float*)d_in[0];
  const float* qIn  = (const float*)d_in[1];
  /* d_in[2] = k : unused by the reference */
  const float* vIn  = (const float*)d_in[3];
  const int*   mask = (const int*)d_in[4];
  const float* Wtau = (const float*)d_in[5];
  const float* btau = (const float*)d_in[6];
  const float* lnvg = (const float*)d_in[7];
  const float* lnvb = (const float*)d_in[8];
  const float* Wv   = (const float*)d_in[9];
  const float* bv   = (const float*)d_in[10];
  const float* Wp   = (const float*)d_in[11];
  const float* bp   = (const float*)d_in[12];
  const float* gpre = (const float*)d_in[13];
  const float* bpre = (const float*)d_in[14];
  const float* Wm1  = (const float*)d_in[15];
  const float* bm1  = (const float*)d_in[16];
  const float* Wm2  = (const float*)d_in[17];
  const float* bm2  = (const float*)d_in[18];
  const float* gpost = (const float*)d_in[19];
  const float* bpost = (const float*)d_in[20];
  float* out = (float*)d_out;

  char* w = (char*)d_ws;
  float*          tauW = (float*)(w);                       // 4096*8*4   = 128KB
  float*          vpW  = (float*)(w + 131072);              // 4096*256*4 = 4MB
  unsigned short* vpT  = (unsigned short*)(w + 4325376);    // 256*4096*2 = 2MB
  unsigned short* aB   = (unsigned short*)(w + 6422528);    // 4096*256*2 = 2MB
  unsigned short* WpB  = (unsigned short*)(w + 8519680);    // 128KB
  unsigned short* Wm1B = (unsigned short*)(w + 8650752);    // 256KB
  unsigned short* Wm2B = (unsigned short*)(w + 8912896);    // 256KB

  prep_kernel<<<QN, 256, 0, stream>>>(vIn, qIn, lnvg, lnvb, Wv, bv, Wtau, btau, vpW, tauW);
  cvtw_kernel<<<1280, 256, 0, stream>>>(Wp, Wm1, Wm2, WpB, Wm1B, Wm2B);
  transpose_kernel<<<dim3(QN / 32, CN / 32), 256, 0, stream>>>(vpW, vpT);
  flash_kernel<<<QN / 16, 256, 0, stream>>>(qpos, mask, tauW, vpT, aB);
  epilogue_kernel<<<QN / 16, 256, 0, stream>>>(aB, WpB, bp, gpre, bpre,
                                               Wm1B, bm1, Wm2B, bm2, gpost, bpost, out);
}
